// GlobalWorkspace_87076166960170
// MI455X (gfx1250) — matrix operand order repaired and checked by reading, not compiled, not run on hardware
//
#include <hip/hip_runtime.h>

// ---------------------------------------------------------------------------
// MI455X (gfx1250, wave32) GlobalWorkspace block.
// GEMMs: bf16 inputs (pre-converted once per launch) via
// v_wmma_f32_16x16x32_bf16 (f32 accumulate). Block tile 128x128, 8 waves
// (4M x 2N), wave tile 32x64 = 8 WMMA per K-step. K-tiles staged into
// double-buffered LDS with GLOBAL_LOAD_ASYNC_TO_LDS_B128 (ASYNCcnt),
// overlapping the tensor-DMA-style copy with WMMA compute.
// ---------------------------------------------------------------------------

typedef __attribute__((ext_vector_type(16))) __bf16 v16bf;
typedef __attribute__((ext_vector_type(8)))  __bf16 v8bf;
typedef __attribute__((ext_vector_type(4)))  __bf16 v4bf;
typedef __attribute__((ext_vector_type(8)))  float  v8f;
typedef __attribute__((ext_vector_type(4)))  float  v4f;

#define LDT   40              // bf16 LDS row stride (80B rows, 16B aligned)
#define TILE  (128 * LDT)     // one 128x32 bf16 tile in LDS
#define SBUF  (2 * TILE)      // A tile + B tile per buffer

__device__ __forceinline__ v16bf mk16(v8bf lo, v8bf hi) {
  v16bf r;
#pragma unroll
  for (int i = 0; i < 8; ++i) { r[i] = lo[i]; r[i + 8] = hi[i]; }
  return r;
}

// Low 32 bits of a flat pointer into LDS == wave-relative LDS byte address
// (AS(3) -> flat addrspacecast places the LDS offset in the low dword).
__device__ __forceinline__ unsigned lds_addr_of(const void* p) {
  return (unsigned)(unsigned long long)p;
}

__device__ __forceinline__ void async_copy_b128(void* lds_dst, const void* gsrc) {
#if defined(__gfx1250__)
  asm volatile("global_load_async_to_lds_b128 %0, %1, off"
               :: "v"(lds_addr_of(lds_dst)),
                  "v"((unsigned long long)gsrc)
               : "memory");
#else
  *(v8bf*)lds_dst = *(const v8bf*)gsrc;
#endif
}

__device__ __forceinline__ void wait_async() {
#if defined(__gfx1250__)
  asm volatile("s_wait_asynccnt 0" ::: "memory");
#endif
}

// ---------------------------------------------------------------------------
// C[M,N] = A[M,K] @ W[N,K]^T + bias[N]   (A,W bf16 row-major; C fp32)
// Branch-free staging: OOB rows are clamped (their products are never stored).
// ---------------------------------------------------------------------------
__global__ __launch_bounds__(256)
void gemm_bf16_wmma(const __bf16* __restrict__ A, const __bf16* __restrict__ W,
                    const float* __restrict__ bias, float* __restrict__ C,
                    int M, int N, int K) {
  __shared__ __bf16 smem[2 * SBUF];     // [buf][A|B][128*LDT]  = 40 KB

  const int tid   = threadIdx.x;
  const int lane  = tid & 31;
  const int wave  = tid >> 5;
  const int waveM = wave & 3;    // 0..3  (rows of 32)
  const int waveN = wave >> 2;   // 0..1  (cols of 64)
  const int bm    = blockIdx.x * 128;
  const int bn    = blockIdx.y * 128;
  const int half  = lane >> 4;   // 0: lanes 0-15, 1: lanes 16-31
  const int l16   = lane & 15;

  // per-thread staging coordinates (2 chunks of 8 bf16 per tile)
  const int r_0 = tid >> 2;              // t=0 row
  const int c_0 = (tid & 3) << 3;        // t=0 col
  const int r_1 = (tid + 256) >> 2;      // t=1 row
  const int c_1 = ((tid + 256) & 3) << 3;
  const int gaA0 = (bm + r_0) < M ? (bm + r_0) : (M - 1);
  const int gaA1 = (bm + r_1) < M ? (bm + r_1) : (M - 1);
  const int gbB0 = (bn + r_0) < N ? (bn + r_0) : (N - 1);
  const int gbB1 = (bn + r_1) < N ? (bn + r_1) : (N - 1);

  auto stage = [&](int buf, int k0) {
    __bf16* As = &smem[buf * SBUF];
    __bf16* Bs = As + TILE;
    async_copy_b128(&As[r_0 * LDT + c_0], A + (size_t)gaA0 * K + k0 + c_0);
    async_copy_b128(&Bs[r_0 * LDT + c_0], W + (size_t)gbB0 * K + k0 + c_0);
    async_copy_b128(&As[r_1 * LDT + c_1], A + (size_t)gaA1 * K + k0 + c_1);
    async_copy_b128(&Bs[r_1 * LDT + c_1], W + (size_t)gbB1 * K + k0 + c_1);
  };

  v8f acc[2][4] = {};

  stage(0, 0);
  wait_async();
  __syncthreads();

  int buf = 0;
  for (int k0 = 0; k0 < K; k0 += 32) {
    const int nbuf = buf ^ 1;
    if (k0 + 32 < K) stage(nbuf, k0 + 32);   // async: overlaps with WMMA below

    const __bf16* As = &smem[buf * SBUF];
    const __bf16* Bs = As + TILE;

    // B fragments: lane l16 = column n; half-wave selects K 0-15 / 16-31
    v16bf bfrag[4];
#pragma unroll
    for (int tj = 0; tj < 4; ++tj) {
      int col = waveN * 64 + tj * 16 + l16;
      const v8bf* p = (const v8bf*)&Bs[col * LDT + half * 16];
      bfrag[tj] = mk16(p[0], p[1]);
    }
    // A fragments: lanes<16 hold K{0..15}, lanes>=16 hold K{16..31}
#pragma unroll
    for (int ti = 0; ti < 2; ++ti) {
      int row = waveM * 32 + ti * 16 + l16;
      int klo = half * 16;
      const v8bf* p0 = (const v8bf*)&As[row * LDT + klo];
      const v8bf* p1 = (const v8bf*)&As[row * LDT + klo + 8];
      v16bf afrag = mk16(p0[0], p1[0]);
#pragma unroll
      for (int tj = 0; tj < 4; ++tj) {
        acc[ti][tj] = __builtin_amdgcn_wmma_f32_16x16x32_bf16(
            false, afrag, false, bfrag[tj], (short)0, acc[ti][tj], false, false);
      }
    }

    wait_async();          // next tile's DMA done (no-op when none issued)
    __syncthreads();       // all waves done reading buf + staging nbuf
    buf = nbuf;
  }

  // store: C/D layout: VGPR j, lanes<16 -> M=j, lanes>=16 -> M=j+8
#pragma unroll
  for (int ti = 0; ti < 2; ++ti) {
#pragma unroll
    for (int tj = 0; tj < 4; ++tj) {
      int n = bn + waveN * 64 + tj * 16 + l16;
      if (n >= N) continue;
      float bv = bias ? bias[n] : 0.0f;
#pragma unroll
      for (int j = 0; j < 8; ++j) {
        int m = bm + waveM * 32 + ti * 16 + j + half * 8;
        if (m < M) C[(size_t)m * N + n] = acc[ti][tj][j] + bv;
      }
    }
  }
}

// ---------------------------------------------------------------------------
// fp32 -> bf16 bulk convert (n multiple of 4)
// ---------------------------------------------------------------------------
__global__ __launch_bounds__(256)
void f32_to_bf16_kernel(const float* __restrict__ in, __bf16* __restrict__ out,
                        long n4) {           // n4 = n/4
  long stride = (long)gridDim.x * 256;
  for (long i = (long)blockIdx.x * 256 + threadIdx.x; i < n4; i += stride) {
    v4f v = *(const v4f*)(in + i * 4);
    v4bf o;
    o[0] = (__bf16)v.x; o[1] = (__bf16)v.y; o[2] = (__bf16)v.z; o[3] = (__bf16)v.w;
    *(v4bf*)(out + i * 4) = o;
  }
}

// ---------------------------------------------------------------------------
// RMS norm, optional residual, optional fp32 and/or bf16 outputs (D == 2048)
// ---------------------------------------------------------------------------
__global__ __launch_bounds__(256)
void rms_kernel(const float* __restrict__ in, const float* __restrict__ res,
                const float* __restrict__ w, float* __restrict__ outf,
                __bf16* __restrict__ outbf, int D) {
  const int row = blockIdx.x;
  const int tid = threadIdx.x;
  const float* x = in + (size_t)row * D;
  const float* r = res ? res + (size_t)row * D : nullptr;
  float v[8];
  float ss = 0.0f;
  const int nper = D / 256;      // 8 for D=2048
#pragma unroll
  for (int i = 0; i < 8; ++i) {
    if (i >= nper) break;
    int d = tid + i * 256;
    float t = x[d];
    if (r) t += r[d];
    v[i] = t;
    ss += t * t;
  }
  __shared__ float red[256];
  red[tid] = ss;
  __syncthreads();
  for (int s = 128; s > 0; s >>= 1) {
    if (tid < s) red[tid] += red[tid + s];
    __syncthreads();
  }
  float scale = rsqrtf(red[0] / (float)D + 1e-6f);
#pragma unroll
  for (int i = 0; i < 8; ++i) {
    if (i >= nper) break;
    int d = tid + i * 256;
    float o = v[i] * scale * w[d];
    if (outf)  outf[(size_t)row * D + d] = o;
    if (outbf) outbf[(size_t)row * D + d] = (__bf16)o;
  }
}

// ---------------------------------------------------------------------------
// per (b,s): top-4 over L of logits[b, :, s], softmax -> wgt, idx
// ---------------------------------------------------------------------------
__global__ __launch_bounds__(256)
void topk_softmax_kernel(const float* __restrict__ logits, int* __restrict__ idx,
                         float* __restrict__ wgt, int L, int S) {
  const int tid = threadIdx.x;
  const int b = blockIdx.x / S;
  const int s = blockIdx.x % S;
  const float* lp = logits + (size_t)b * L * S + s;

  float tv[4] = {-3.0e38f, -3.0e38f, -3.0e38f, -3.0e38f};
  int   ti[4] = {0, 0, 0, 0};
  for (int l = tid; l < L; l += 256) {
    float val = lp[(size_t)l * S];
    if (val > tv[3]) {
      int p = 3;
      while (p > 0 && val > tv[p - 1]) { tv[p] = tv[p - 1]; ti[p] = ti[p - 1]; --p; }
      tv[p] = val; ti[p] = l;
    }
  }
  __shared__ float sv[256][4];
  __shared__ int   si[256][4];
#pragma unroll
  for (int i = 0; i < 4; ++i) { sv[tid][i] = tv[i]; si[tid][i] = ti[i]; }
  __syncthreads();
  for (int stride = 128; stride > 0; stride >>= 1) {
    if (tid < stride) {
      float av[4]; int ai[4];
      int pa = 0, pb = 0;
#pragma unroll
      for (int o = 0; o < 4; ++o) {
        float va = sv[tid][pa], vb = sv[tid + stride][pb];
        if (va >= vb) { av[o] = va; ai[o] = si[tid][pa]; ++pa; }
        else          { av[o] = vb; ai[o] = si[tid + stride][pb]; ++pb; }
      }
#pragma unroll
      for (int o = 0; o < 4; ++o) { sv[tid][o] = av[o]; si[tid][o] = ai[o]; }
    }
    __syncthreads();
  }
  if (tid == 0) {
    float mx = sv[0][0];
    float e[4], sum = 0.0f;
#pragma unroll
    for (int i = 0; i < 4; ++i) { e[i] = __expf(sv[0][i] - mx); sum += e[i]; }
#pragma unroll
    for (int i = 0; i < 4; ++i) {
      wgt[blockIdx.x * 4 + i] = e[i] / sum;
      idx[blockIdx.x * 4 + i] = si[0][i];
    }
  }
}

// ---------------------------------------------------------------------------
// out[b,s,:] = ws[b,s,:] + sum_k wgt[b,s,k] * wx[b, idx[b,s,k], :]
// ---------------------------------------------------------------------------
__global__ __launch_bounds__(256)
void gather_write_kernel(const float* __restrict__ wx, const int* __restrict__ idx,
                         const float* __restrict__ wgt, const float* __restrict__ ws,
                         float* __restrict__ out, int L, int D, int S) {
  const int bs = blockIdx.x;
  const int b = bs / S;
  const int* ip = idx + bs * 4;
  const float* wp = wgt + bs * 4;
  int i0 = ip[0], i1 = ip[1], i2 = ip[2], i3 = ip[3];
  float w0 = wp[0], w1 = wp[1], w2 = wp[2], w3 = wp[3];
  for (int d = threadIdx.x; d < D; d += 256) {
    float acc = ws[(size_t)bs * D + d];
    acc += w0 * wx[((size_t)b * L + i0) * D + d];
    acc += w1 * wx[((size_t)b * L + i1) * D + d];
    acc += w2 * wx[((size_t)b * L + i2) * D + d];
    acc += w3 * wx[((size_t)b * L + i3) * D + d];
    out[(size_t)bs * D + d] = acc;
  }
}

// ---------------------------------------------------------------------------
// Attention: one wave32 per (b,l,h); S=16 slots on lanes 0..15 (upper half
// mirrors); hd=64 -> 2 output dims per lane. Writes bf16 ctx directly.
// ---------------------------------------------------------------------------
__global__ __launch_bounds__(256)
void attn_kernel(const float* __restrict__ q, const float* __restrict__ kk,
                 const float* __restrict__ vv, __bf16* __restrict__ ctx,
                 int B, int L, int H, int S) {
  const int hd = 64;
  const int lane = threadIdx.x & 31;
  const int wid = (blockIdx.x * blockDim.x + threadIdx.x) >> 5;
  const int total = B * L * H;
  if (wid >= total) return;
  const int h = wid % H;
  const int bl = wid / H;
  const int l = bl % L;
  const int b = bl / L;

  const float* qp = q + ((size_t)b * L + l) * (size_t)H * hd + (size_t)h * hd;
  const int s = lane & 15;
  const float* ks = kk + ((size_t)b * S + s) * (size_t)H * hd + (size_t)h * hd;
  float sc = 0.0f;
#pragma unroll
  for (int d = 0; d < hd; d += 4) {
    v4f qv = *(const v4f*)(qp + d);
    v4f kv = *(const v4f*)(ks + d);
    sc += qv.x * kv.x + qv.y * kv.y + qv.z * kv.z + qv.w * kv.w;
  }
  sc *= 0.125f;                                    // 1/sqrt(64)

  float m = sc;
  for (int o = 8; o > 0; o >>= 1) m = fmaxf(m, __shfl_xor(m, o));
  float e = __expf(sc - m);
  float sum = e;
  for (int o = 8; o > 0; o >>= 1) sum += __shfl_xor(sum, o);
  float att = e / sum;

  const float* vp = vv + (size_t)b * S * (size_t)H * hd + (size_t)h * hd;
  const int d0 = lane * 2;
  float a0 = 0.0f, a1 = 0.0f;
#pragma unroll
  for (int ss = 0; ss < 16; ++ss) {
    float a = __shfl(att, ss);
    const float* vs = vp + (size_t)ss * H * hd;
    a0 += a * vs[d0];
    a1 += a * vs[d0 + 1];
  }
  __bf16* cp = ctx + ((size_t)b * L + l) * (size_t)H * hd + (size_t)h * hd;
  cp[d0]     = (__bf16)a0;
  cp[d0 + 1] = (__bf16)a1;
}

// ---------------------------------------------------------------------------
// Host orchestration (graph-capture safe)
// ---------------------------------------------------------------------------
extern "C" void kernel_launch(void* const* d_in, const int* in_sizes, int n_in,
                              void* d_out, int out_size, void* d_ws, size_t ws_size,
                              hipStream_t stream) {
  (void)in_sizes; (void)n_in; (void)out_size; (void)ws_size;
  const int B = 4, L = 2048, D = 2048, S = 16, H = 32, NB = 2;
  const size_t BIG = (size_t)B * L * D;            // 16.78M elements

  const float* x_in       = (const float*)d_in[0];
  const float* workspace  = (const float*)d_in[1];
  const float* compete_w  = (const float*)d_in[2];
  const float* compete_b  = (const float*)d_in[3];
  const float* write_w    = (const float*)d_in[4];
  const float* write_b    = (const float*)d_in[5];
  const float* in_proj_w  = (const float*)d_in[6];
  const float* in_proj_b  = (const float*)d_in[7];
  const float* out_proj_w = (const float*)d_in[8];
  const float* out_proj_b = (const float*)d_in[9];
  const float* norm_pre_w = (const float*)d_in[12];
  const float* norm_post_w= (const float*)d_in[13];

  // ---- workspace carve-up (byte offsets, 256B aligned) ----
  char* base = (char*)d_ws;
  size_t off = 0;
  auto carve = [&](size_t bytes) {
    char* p = base + off;
    off += (bytes + 255) & ~(size_t)255;
    return p;
  };
  float*  X       = (float*) carve(BIG * 4);            // residual stream (fp32)
  float*  Tbuf    = (float*) carve(BIG * 4);            // wx -> q -> xb (fp32)
  float*  logits  = (float*) carve((size_t)B * L * S * 4);
  float*  ws      = (float*) carve((size_t)B * S * D * 4);
  float*  wsin    = (float*) carve((size_t)B * S * D * 4);
  float*  kkb     = (float*) carve((size_t)B * S * D * 4);
  float*  vvb     = (float*) carve((size_t)B * S * D * 4);
  float*  wgt     = (float*) carve((size_t)B * S * 4 * 4);
  int*    idx     = (int*)   carve((size_t)B * S * 4 * 4);
  __bf16* Xbf     = (__bf16*)carve(BIG * 2);            // bf16 shadow of X
  __bf16* Ubf     = (__bf16*)carve(BIG * 2);            // xn -> ctx (bf16)
  __bf16* wsbf    = (__bf16*)carve((size_t)B * S * D * 2);
  __bf16* competeBf = (__bf16*)carve((size_t)S * D * 2);
  __bf16* writeBf   = (__bf16*)carve((size_t)D * D * 2);
  __bf16* inprojBf  = (__bf16*)carve((size_t)3 * D * D * 2);
  __bf16* outBf     = (__bf16*)carve((size_t)D * D * 2);

  const __bf16* wqBf = inprojBf;
  const __bf16* wkBf = inprojBf + (size_t)D * D;
  const __bf16* wvBf = inprojBf + (size_t)2 * D * D;
  const float* bq = in_proj_b;
  const float* bk = in_proj_b + D;
  const float* bv = in_proj_b + 2 * D;

  const dim3 blk(256);
  const dim3 gFull(64, 16);      // M=8192/128, N=2048/128
  const dim3 gCompete(64, 1);    // N=16 -> single (mostly idle) N tile
  const dim3 gWs(1, 16);         // M=64 -> single M tile

  // ---- once-per-launch setup: residual copy + bf16 weight/activation prep ----
  hipMemcpyAsync(X, x_in, BIG * sizeof(float), hipMemcpyDeviceToDevice, stream);
  for (int b = 0; b < B; ++b)
    hipMemcpyAsync(ws + (size_t)b * S * D, workspace, (size_t)S * D * sizeof(float),
                   hipMemcpyDeviceToDevice, stream);
  f32_to_bf16_kernel<<<2048, blk, 0, stream>>>(x_in, Xbf, (long)(BIG / 4));
  f32_to_bf16_kernel<<<64,   blk, 0, stream>>>(ws, wsbf, (long)(B * S * D / 4));
  f32_to_bf16_kernel<<<32,   blk, 0, stream>>>(compete_w, competeBf, (long)(S * D / 4));
  f32_to_bf16_kernel<<<1024, blk, 0, stream>>>(write_w, writeBf, (long)((size_t)D * D / 4));
  f32_to_bf16_kernel<<<2048, blk, 0, stream>>>(in_proj_w, inprojBf, (long)((size_t)3 * D * D / 4));
  f32_to_bf16_kernel<<<1024, blk, 0, stream>>>(out_proj_w, outBf, (long)((size_t)D * D / 4));

  for (int it = 0; it < NB; ++it) {
    // 1. xn = rms(x, norm_pre_w)  (bf16 only; feeds compete GEMM)
    rms_kernel<<<B * L, blk, 0, stream>>>(X, nullptr, norm_pre_w, nullptr, Ubf, D);
    // 2. logits = xn @ compete_w^T + compete_b
    gemm_bf16_wmma<<<gCompete, blk, 0, stream>>>(Ubf, competeBf, compete_b, logits,
                                                 B * L, S, D);
    // 3. per-slot top-4 + softmax
    topk_softmax_kernel<<<B * S, blk, 0, stream>>>(logits, idx, wgt, L, S);
    // 4. wx = x @ write_w^T + write_b
    gemm_bf16_wmma<<<gFull, blk, 0, stream>>>(Xbf, writeBf, write_b, Tbuf, B * L, D, D);
    // 5. ws = rms(ws + gathered-write, norm_post_w)  (fp32 + bf16)
    gather_write_kernel<<<B * S, blk, 0, stream>>>(Tbuf, idx, wgt, ws, wsin, L, D, S);
    rms_kernel<<<B * S, blk, 0, stream>>>(wsin, nullptr, norm_post_w, ws, wsbf, D);
    // 6. q = x @ wq^T + bq
    gemm_bf16_wmma<<<gFull, blk, 0, stream>>>(Xbf, wqBf, bq, Tbuf, B * L, D, D);
    // 7. kk/vv = ws @ wk^T / wv^T
    gemm_bf16_wmma<<<gWs, blk, 0, stream>>>(wsbf, wkBf, bk, kkb, B * S, D, D);
    gemm_bf16_wmma<<<gWs, blk, 0, stream>>>(wsbf, wvBf, bv, vvb, B * S, D, D);
    // 8. attention -> ctx (bf16)
    attn_kernel<<<(B * L * H) / 8, blk, 0, stream>>>(Tbuf, kkb, vvb, Ubf, B, L, H, S);
    // 9. xb = ctx @ out_proj^T + out_proj_b
    gemm_bf16_wmma<<<gFull, blk, 0, stream>>>(Ubf, outBf, out_proj_b, Tbuf, B * L, D, D);
    // 10. x = rms(x + xb, norm_post_w)  (fp32 X + bf16 shadow; ignition is dead code)
    rms_kernel<<<B * L, blk, 0, stream>>>(X, Tbuf, norm_post_w, X, Xbf, D);
  }

  hipMemcpyAsync(d_out, X, BIG * sizeof(float), hipMemcpyDeviceToDevice, stream);
}
